// CoSentClusteringLoss_30485677867129
// MI455X (gfx1250) — matrix operand order repaired and checked by reading, not compiled, not run on hardware
//
#include <hip/hip_runtime.h>
#include <hip/hip_bf16.h>
#include <math.h>

typedef __attribute__((ext_vector_type(16))) _Float16 v16h;
typedef __attribute__((ext_vector_type(8)))  float    v8f;
typedef __attribute__((ext_vector_type(4)))  float    v4f;

#define N_ 8192
#define D_ 256
#define NUM_LABELS_ 128

union F16Frag {            // 32 bytes = 8 VGPRs = one WMMA f16 A/B fragment per lane
    v16h h;
    v4f  q[2];
};

// ---------------------------------------------------------------------------
// Stage 1: row-wise L2 normalize, convert to f16 (WMMA operand precision),
// and zero the per-row accumulators a[], b[] (workspace is poisoned).
// One block (256 threads) per row; D == 256 so one element per thread.
// ---------------------------------------------------------------------------
__global__ void __launch_bounds__(256)
cosent_normalize_f16(const float* __restrict__ emb, _Float16* __restrict__ eh,
                     float* __restrict__ a, float* __restrict__ b) {
    __shared__ float red[8];
    const int row  = blockIdx.x;
    const int d    = threadIdx.x;
    const int lane = threadIdx.x & 31;
    const int wid  = threadIdx.x >> 5;

    float x  = emb[row * D_ + d];
    float ss = x * x;
    #pragma unroll
    for (int m = 16; m >= 1; m >>= 1) ss += __shfl_xor(ss, m, 32);
    if (lane == 0) red[wid] = ss;
    __syncthreads();
    if (wid == 0) {
        float v = (lane < 8) ? red[lane] : 0.0f;
        #pragma unroll
        for (int m = 4; m >= 1; m >>= 1) v += __shfl_xor(v, m, 32);
        if (lane == 0) red[0] = v;
    }
    __syncthreads();
    const float norm = fmaxf(sqrtf(red[0]), 1e-12f);   // matches reference eps clamp
    eh[row * D_ + d] = (_Float16)(x / norm);
    if (d == 0) { a[row] = 0.0f; b[row] = 0.0f; }
}

// ---------------------------------------------------------------------------
// Stage 2: fused WMMA Gram tile + exp/mask + row reduction.
// Each wave owns one 16-row i-band and sweeps 16 j-tiles (one 256-col chunk).
// Grid: (512/8 bands, 32 chunks), block = 256 threads = 8 waves.
// S is never written to memory: per-tile f32 accumulators are consumed by
// v_exp_f32 immediately and folded into per-row a/b partial sums.
// ---------------------------------------------------------------------------
__global__ void __launch_bounds__(256)
cosent_gram_wmma(const _Float16* __restrict__ eh, const int* __restrict__ labels,
                 const float* __restrict__ logit_scale,
                 float* __restrict__ a, float* __restrict__ b) {
    const float s    = logit_scale[0];
    const int   lane = threadIdx.x & 31;
    const int   wave = threadIdx.x >> 5;
    const int   i0   = (blockIdx.x * 8 + wave) * 16;   // i-band base row
    const int   half = lane >> 4;                      // 0: lanes 0-15, 1: lanes 16-31
    const int   mrow = lane & 15;

    // --- Preload A fragments for this i-band (ISA 16-bit A 16x32 layout):
    //     lane half=0 -> K {0..7} U {16..23}; half=1 -> K {8..15} U {24..31}.
    v16h afr[8];
    {
        const _Float16* arow = eh + (size_t)(i0 + mrow) * D_;
        #pragma unroll
        for (int kb = 0; kb < 8; ++kb) {
            F16Frag f;
            const int k0 = kb * 32 + half * 8;
            f.q[0] = *(const v4f*)(arow + k0);
            f.q[1] = *(const v4f*)(arow + k0 + 16);
            afr[kb] = f.h;
        }
    }

    // Row labels for the 8 C/D VGPR rows this lane-half produces.
    int li[8];
    #pragma unroll
    for (int r = 0; r < 8; ++r) li[r] = labels[i0 + half * 8 + r];

    float aacc[8], bacc[8];
    #pragma unroll
    for (int r = 0; r < 8; ++r) { aacc[r] = 0.0f; bacc[r] = 0.0f; }

    const int jt0 = blockIdx.y * 16;                   // first j-tile of this chunk
    for (int t = 0; t < 16; ++t) {
        const int j0 = (jt0 + t) * 16;
        const _Float16* brow = eh + (size_t)(j0 + mrow) * D_;

        v8f c = {};
        #pragma unroll
        for (int kb = 0; kb < 8; ++kb) {
            // ISA 16-bit B 32x16 layout: lanes 0-15 hold K {0..7} U {16..23}, lanes 16-31 K {8..15} U {24..31}
            F16Frag bf;
            const int k0 = kb * 32 + half * 8;
            bf.q[0] = *(const v4f*)(brow + k0);
            bf.q[1] = *(const v4f*)(brow + k0 + 16);
            c = __builtin_amdgcn_wmma_f32_16x16x32_f16(
                    /*neg_a=*/false, afr[kb], /*neg_b=*/false, bf.h,
                    /*c_mod=*/(short)0, c, /*reuse_a=*/false, /*reuse_b=*/false);
        }

        // C/D layout: VGPR r, lanes 0-15 -> (M=r, N=lane); lanes 16-31 -> (M=8+r, N=lane-16)
        const int jg = j0 + mrow;
        const int lj = labels[jg];
        #pragma unroll
        for (int r = 0; r < 8; ++r) {
            const int   ig   = i0 + half * 8 + r;
            const float Sv   = c[r] * s;
            const bool  same = (li[r] == lj);
            const bool  diag = (ig == jg);
            aacc[r] += (same && !diag) ? __expf(-Sv) : 0.0f;
            bacc[r] += (!same)         ? __expf( Sv) : 0.0f;
        }
    }

    // Reduce across the 16 columns (within each half-wave), then one atomic per row.
    #pragma unroll
    for (int r = 0; r < 8; ++r) {
        float av = aacc[r], bv = bacc[r];
        #pragma unroll
        for (int m = 8; m >= 1; m >>= 1) {
            av += __shfl_xor(av, m, 32);
            bv += __shfl_xor(bv, m, 32);
        }
        if (mrow == 0) {
            const int ig = i0 + half * 8 + r;
            atomicAdd(&a[ig], av);
            atomicAdd(&b[ig], bv);
        }
    }
}

// ---------------------------------------------------------------------------
// Stage 3: segment-sum a/b/counts per label in LDS, then log(1 + sum(A*B)).
// Single block; 8192 rows / 128 labels is trivial work.
// ---------------------------------------------------------------------------
__global__ void __launch_bounds__(256)
cosent_finalize(const float* __restrict__ a, const float* __restrict__ b,
                const int* __restrict__ labels, float* __restrict__ out) {
    __shared__ float As[NUM_LABELS_];
    __shared__ float Bs[NUM_LABELS_];
    __shared__ int   Cs[NUM_LABELS_];
    __shared__ float rsum[8];
    const int tid = threadIdx.x;

    if (tid < NUM_LABELS_) { As[tid] = 0.0f; Bs[tid] = 0.0f; Cs[tid] = 0; }
    __syncthreads();

    for (int r = tid; r < N_; r += 256) {
        const int g = labels[r];
        atomicAdd(&As[g], a[r]);
        atomicAdd(&Bs[g], b[r]);
        atomicAdd(&Cs[g], 1);
    }
    __syncthreads();

    float v = 0.0f;
    if (tid < NUM_LABELS_ && Cs[tid] >= 2) v = As[tid] * Bs[tid];
    #pragma unroll
    for (int m = 16; m >= 1; m >>= 1) v += __shfl_xor(v, m, 32);
    if ((tid & 31) == 0) rsum[tid >> 5] = v;
    __syncthreads();
    if (tid == 0) {
        float t = 0.0f;
        #pragma unroll
        for (int w = 0; w < 8; ++w) t += rsum[w];
        out[0] = logf(1.0f + t);
    }
}

extern "C" void kernel_launch(void* const* d_in, const int* in_sizes, int n_in,
                              void* d_out, int out_size, void* d_ws, size_t ws_size,
                              hipStream_t stream) {
    const float* emb    = (const float*)d_in[0];
    const int*   labels = (const int*)d_in[1];
    const float* lscale = (const float*)d_in[2];

    char*      ws = (char*)d_ws;
    _Float16*  eh = (_Float16*)ws;                         // 8192*256*2 = 4 MB
    float*     a  = (float*)(ws + (size_t)N_ * D_ * 2);    // 32 KB
    float*     b  = a + N_;                                // 32 KB
    float*     out = (float*)d_out;

    cosent_normalize_f16<<<N_, 256, 0, stream>>>(emb, eh, a, b);

    dim3 grid(64, 32);                                     // 512 i-bands / 8 waves, 32 j-chunks
    cosent_gram_wmma<<<grid, 256, 0, stream>>>(eh, labels, lscale, a, b);

    cosent_finalize<<<1, 256, 0, stream>>>(a, b, labels, out);
}
